// LSTMModel_3238405341925
// MI455X (gfx1250) — matrix operand order repaired and checked by reading, not compiled, not run on hardware
//
#include <hip/hip_runtime.h>

typedef __bf16 bf16;
typedef __attribute__((ext_vector_type(16))) __bf16 v16bf;
typedef __attribute__((ext_vector_type(8)))  __bf16 v8bf;
typedef __attribute__((ext_vector_type(8)))  float  v8f;

// Problem sizes
#define E_SZ 300
#define EP   320          // E padded to multiple of 32 (WMMA K)
#define H_SZ 512
#define G4   2048         // 4*H
#define B_SZ 256
#define S_SZ 256
#define KC   (EP + H_SZ)  // 832 combined K per step

// Workspace layout (bytes)
static constexpr size_t OFF_EMB = 0;                                   // bf16 [S][B][EP]
static constexpr size_t SZ_EMB  = (size_t)S_SZ * B_SZ * EP * 2;        // 41,943,040
static constexpr size_t OFF_WF  = OFF_EMB + SZ_EMB;                    // bf16 [2048][832]
static constexpr size_t SZ_W    = (size_t)G4 * KC * 2;                 //  3,407,872
static constexpr size_t OFF_WB  = OFF_WF + SZ_W;
static constexpr size_t OFF_HF  = OFF_WB + SZ_W;                       // bf16 [2][B][H]
static constexpr size_t SZ_H    = (size_t)2 * B_SZ * H_SZ * 2;         //    524,288
static constexpr size_t OFF_HB  = OFF_HF + SZ_H;
static constexpr size_t OFF_CF  = OFF_HB + SZ_H;                       // f32 [B][H]
static constexpr size_t SZ_C    = (size_t)B_SZ * H_SZ * 4;             //    524,288
static constexpr size_t OFF_CB  = OFF_CF + SZ_C;
// zero region: h_f, h_b, c_f, c_b  (contiguous 2,097,152 B = 524,288 u32)

__device__ __forceinline__ float sigm(float x)  { return 1.0f / (1.0f + __expf(-x)); }
__device__ __forceinline__ float tanhf_(float x){ return 2.0f * sigm(2.0f * x) - 1.0f; }

// ---------------- prep kernels ----------------

__global__ void zero_ws_k(unsigned int* __restrict__ p, int n) {
    int i = blockIdx.x * blockDim.x + threadIdx.x;
    if (i < n) p[i] = 0u;
}

// Pack [W_ih | 0-pad | W_hh] into bf16 [2048][832] per direction.
__global__ void build_w_k(const float* __restrict__ Wih_f, const float* __restrict__ Whh_f,
                          const float* __restrict__ Wih_b, const float* __restrict__ Whh_b,
                          bf16* __restrict__ Wc_f, bf16* __restrict__ Wc_b) {
    const int n   = blockIdx.x;
    const int dir = blockIdx.y;
    const float* Wih = dir ? Wih_b : Wih_f;
    const float* Whh = dir ? Whh_b : Whh_f;
    bf16* Wc = dir ? Wc_b : Wc_f;
    for (int k = threadIdx.x; k < KC; k += blockDim.x) {
        float v;
        if (k < E_SZ)      v = Wih[(size_t)n * E_SZ + k];
        else if (k < EP)   v = 0.0f;
        else               v = Whh[(size_t)n * H_SZ + (k - EP)];
        Wc[(size_t)n * KC + k] = (bf16)v;
    }
}

// Gather embeddings into bf16 [S][B][EP] (time-major), pad E 300->320 with zeros.
__global__ void embed_k(const int* __restrict__ x, const float* __restrict__ emb,
                        bf16* __restrict__ embA) {
    const int row = blockIdx.x;            // row = t*B + b
    const int t   = row >> 8;
    const int b   = row & (B_SZ - 1);
    const int tok = x[b * S_SZ + t];       // x is (B,1,S)
    const int e   = threadIdx.x;           // blockDim.x == EP
    float v = (e < E_SZ) ? emb[(size_t)tok * E_SZ + e] : 0.0f;
    embA[(size_t)row * EP + e] = (bf16)v;
}

// ---------------- recurrence step ----------------
// One launch per timestep, both directions. 1024 waves: dir(2) x Mtiles(16) x Ntiles(32).
// Each wave: 16x16 hidden tile; 4 accumulators = gates i/f/g/o; K-loop over [x_t ; h_prev].
__global__ __launch_bounds__(256) void lstm_step_k(
    const bf16* __restrict__ embA,
    const bf16* __restrict__ Wc_f, const bf16* __restrict__ Wc_b,
    const float* __restrict__ bias_f, const float* __restrict__ bias_b,
    bf16* __restrict__ h_f, bf16* __restrict__ h_b,
    float* __restrict__ c_f, float* __restrict__ c_b,
    int t) {
    const int lane  = threadIdx.x & 31;
    const int wave  = threadIdx.x >> 5;
    const int gw    = blockIdx.x * 8 + wave;     // 0..1023
    const int dir   = gw >> 9;
    const int rem   = gw & 511;
    const int mt    = rem >> 5;                  // 0..15 (batch tiles)
    const int nt    = rem & 31;                  // 0..31 (hidden tiles)

    const int lhalf = lane >> 4;                 // 0|1
    const int l15   = lane & 15;

    const bf16*  Wc   = dir ? Wc_b   : Wc_f;
    const float* bias = dir ? bias_b : bias_f;
    bf16*  hbuf = dir ? h_b : h_f;
    float* cbuf = dir ? c_b : c_f;

    const bf16* hprev = hbuf + (size_t)(t & 1)       * (B_SZ * H_SZ);
    bf16*       hnext = hbuf + (size_t)((t + 1) & 1) * (B_SZ * H_SZ);

    const int t_eff = dir ? (S_SZ - 1 - t) : t;

    const int brow = mt * 16 + l15;              // A row (batch index) for this lane
    const int ncol = nt * 16 + l15;              // B column (hidden index) for this lane

    const bf16* aXbase = embA  + ((size_t)t_eff * B_SZ + brow) * EP;
    const bf16* aHbase = hprev + (size_t)brow * H_SZ;

    v8f acc0 = {}, acc1 = {}, acc2 = {}, acc3 = {};

    for (int kk = 0; kk < KC / 32; ++kk) {
        const int k0 = kk * 32;
        // A fragment (contiguous K): lanes 0-15 K{0..15}, lanes 16-31 K{16..31}
        const bf16* arow = (k0 < EP) ? (aXbase + k0) : (aHbase + (k0 - EP));
        v8bf alo = *(const v8bf*)(arow + lhalf * 16);
        v8bf ahi = *(const v8bf*)(arow + lhalf * 16 + 8);
        v16bf afrag = __builtin_shufflevector(alo, ahi,
            0, 1, 2, 3, 4, 5, 6, 7, 8, 9, 10, 11, 12, 13, 14, 15);

        // B fragments, column-major per lane: lane holds col (l&15), 16 contiguous K
        const bf16* wb = Wc + (size_t)ncol * KC + k0 + lhalf * 16;
        v16bf b0 = *(const v16bf*)(wb);                         // gate i
        v16bf b1 = *(const v16bf*)(wb + (size_t)512  * KC);     // gate f
        v16bf b2 = *(const v16bf*)(wb + (size_t)1024 * KC);     // gate g
        v16bf b3 = *(const v16bf*)(wb + (size_t)1536 * KC);     // gate o

        acc0 = __builtin_amdgcn_wmma_f32_16x16x32_bf16(false, afrag, false, b0, (short)0, acc0, false, false);
        acc1 = __builtin_amdgcn_wmma_f32_16x16x32_bf16(false, afrag, false, b1, (short)0, acc1, false, false);
        acc2 = __builtin_amdgcn_wmma_f32_16x16x32_bf16(false, afrag, false, b2, (short)0, acc2, false, false);
        acc3 = __builtin_amdgcn_wmma_f32_16x16x32_bf16(false, afrag, false, b3, (short)0, acc3, false, false);
    }

    // Gate math. C/D layout: VGPR v, lane l -> row mt*16 + v + 8*lhalf, col nt*16 + (l&15).
    const int   hglob = ncol;
    const float bi = bias[0 * H_SZ + hglob];
    const float bf = bias[1 * H_SZ + hglob];
    const float bg = bias[2 * H_SZ + hglob];
    const float bo = bias[3 * H_SZ + hglob];
    const int   bbase = mt * 16 + lhalf * 8;

#pragma unroll
    for (int v = 0; v < 8; ++v) {
        const int    bidx = bbase + v;
        const size_t off  = (size_t)bidx * H_SZ + hglob;
        const float gi = acc0[v] + bi;
        const float gf = acc1[v] + bf;
        const float gg = acc2[v] + bg;
        const float go = acc3[v] + bo;
        const float cold = cbuf[off];
        const float cnew = sigm(gf) * cold + sigm(gi) * tanhf_(gg);
        const float hnew = sigm(go) * tanhf_(cnew);
        cbuf[off]  = cnew;
        hnext[off] = (bf16)hnew;
    }
}

// ---------------- output ----------------
__global__ void out_k(const bf16* __restrict__ hf, const bf16* __restrict__ hb,
                      const float* __restrict__ W_out, const float* __restrict__ b_out,
                      float* __restrict__ out) {
    const int b = threadIdx.x;  // 256 threads
    float acc = 0.0f;
    for (int h = 0; h < H_SZ; ++h) {
        acc += (float)hf[(size_t)b * H_SZ + h] * W_out[h];
        acc += (float)hb[(size_t)b * H_SZ + h] * W_out[H_SZ + h];
    }
    out[b] = acc + b_out[0];
}

extern "C" void kernel_launch(void* const* d_in, const int* in_sizes, int n_in,
                              void* d_out, int out_size, void* d_ws, size_t ws_size,
                              hipStream_t stream) {
    (void)in_sizes; (void)n_in; (void)out_size; (void)ws_size;
    const int*   x      = (const int*)  d_in[0];
    const float* emb    = (const float*)d_in[1];
    const float* Wih_f  = (const float*)d_in[2];
    const float* Whh_f  = (const float*)d_in[3];
    const float* bias_f = (const float*)d_in[4];
    const float* Wih_b  = (const float*)d_in[5];
    const float* Whh_b  = (const float*)d_in[6];
    const float* bias_b = (const float*)d_in[7];
    const float* W_out  = (const float*)d_in[8];
    const float* b_out  = (const float*)d_in[9];
    float* out = (float*)d_out;

    char* ws = (char*)d_ws;
    bf16*  embA = (bf16*)(ws + OFF_EMB);
    bf16*  Wc_f = (bf16*)(ws + OFF_WF);
    bf16*  Wc_b = (bf16*)(ws + OFF_WB);
    bf16*  h_f  = (bf16*)(ws + OFF_HF);
    bf16*  h_b  = (bf16*)(ws + OFF_HB);
    float* c_f  = (float*)(ws + OFF_CF);
    float* c_b  = (float*)(ws + OFF_CB);

    // zero h (both slots, both dirs) and c (both dirs): 2 MiB = 524288 u32
    zero_ws_k<<<2048, 256, 0, stream>>>((unsigned int*)(ws + OFF_HF), 524288);
    // pack weights to bf16 [4H][EP+H]
    build_w_k<<<dim3(G4, 2), 256, 0, stream>>>(Wih_f, Whh_f, Wih_b, Whh_b, Wc_f, Wc_b);
    // embedding gather -> bf16 time-major, E padded
    embed_k<<<S_SZ * B_SZ, EP, 0, stream>>>(x, emb, embA);
    // sequential recurrence, both directions per launch
    for (int t = 0; t < S_SZ; ++t) {
        lstm_step_k<<<128, 256, 0, stream>>>(embA, Wc_f, Wc_b, bias_f, bias_b,
                                             h_f, h_b, c_f, c_b, t);
    }
    // final hidden states live in slot 0 after 256 steps
    out_k<<<1, 256, 0, stream>>>(h_f, h_b, W_out, b_out, out);
}
